// HGAT_90787018703386
// MI455X (gfx1250) — matrix operand order repaired and checked by reading, not compiled, not run on hardware
//
#include <hip/hip_runtime.h>

// ---------------------------------------------------------------------------
// Types for CDNA5 WMMA (wave32): bf16 A/B fragments (16 x bf16 = 8 VGPRs),
// f32 C/D accumulator (8 VGPRs).
// ---------------------------------------------------------------------------
typedef __attribute__((ext_vector_type(16))) __bf16       v16bf;
typedef __attribute__((ext_vector_type(8)))  float        v8f;
typedef __attribute__((ext_vector_type(8)))  unsigned int v8u;

__device__ __forceinline__ unsigned short f2bf(float f) {
  // round-to-nearest-even f32 -> bf16
  unsigned int u = __float_as_uint(f);
  u += 0x7fffu + ((u >> 16) & 1u);
  return (unsigned short)(u >> 16);
}

__device__ __forceinline__ void atomicMaxF(float* addr, float val) {
  // IEEE float max via integer ordering trick (valid incl. -inf init)
  if (val >= 0.0f) atomicMax((int*)addr, __float_as_int(val));
  else             atomicMin((unsigned int*)addr, __float_as_uint(val));
}

// ---------------------------------------------------------------------------
// Pack f32 row-major (M x K) into bf16 A-fragment order.
// Fragment block = (m_tile, k_step): 32 lanes x 16 bf16 (32B/lane, contiguous).
// A 16-bit 16x32 layout (ISA 7.12.2): lane l -> m = l&15, half = l>>4;
//   elems 0..7  : k = k0 + half*8 + e
//   elems 8..15 : k = k0 + 16 + half*8 + (e-8)
// ---------------------------------------------------------------------------
__global__ void pack_a_kernel(const float* __restrict__ X,
                              unsigned short* __restrict__ Ap,
                              int K, int total) {
  int t = blockIdx.x * blockDim.x + threadIdx.x;
  if (t >= total) return;
  int lane = t & 31;
  int rest = t >> 5;
  int Ksteps = K >> 5;
  int ks = rest % Ksteps;
  int mt = rest / Ksteps;
  int m = lane & 15, half = lane >> 4;
  const float* row = X + (size_t)(mt * 16 + m) * K + ks * 32;
  unsigned short* o = Ap + (size_t)t * 16;
#pragma unroll
  for (int e = 0; e < 8; ++e) o[e]     = f2bf(row[half * 8 + e]);
#pragma unroll
  for (int e = 0; e < 8; ++e) o[8 + e] = f2bf(row[16 + half * 8 + e]);
}

// ---------------------------------------------------------------------------
// Pack W (K x 128, f32 row-major) into bf16 B-fragment order.
// Fragment block = (k_step, n_tile): lane l -> n = l&15, half = l>>4;
//   elems 0..7 : k = k0 + half*8 + e; elems 8..15 : k = k0 + 16 + half*8 + (e-8)
// ---------------------------------------------------------------------------
__global__ void pack_b_kernel(const float* __restrict__ W,
                              unsigned short* __restrict__ Bp,
                              int K, int total) {
  int t = blockIdx.x * blockDim.x + threadIdx.x;
  if (t >= total) return;
  int lane = t & 31;
  int rest = t >> 5;
  int nt = rest & 7;
  int ks = rest >> 3;
  int n = lane & 15, half = lane >> 4;
  unsigned short* o = Bp + (size_t)t * 16;
#pragma unroll
  for (int e = 0; e < 16; ++e)
    o[e] = f2bf(W[(size_t)(ks * 32 + (e >> 3) * 16 + half * 8 + (e & 7)) * 128 + nt * 16 + n]);
}

// ---------------------------------------------------------------------------
// C[M x 128] = Apack(M x K, bf16) * Bpack(K x 128, bf16), f32 accumulate.
// Block = 256 threads = 8 wave32s = 4 m-tiles x 2 n-halves.
// Each wave owns a 16x64 C tile (4 accumulators): one A fragment feeds 4
// WMMAs -> 10 b128 loads per 4 v_wmma (vs 16 for a 16x16 tile).
// Out-of-range m-tiles clamp (duplicate identical writes) so EXEC stays
// all-ones for every WMMA (ISA 7.12 requirement).
// ---------------------------------------------------------------------------
__global__ void __launch_bounds__(256)
gemm_wmma_kernel(const unsigned short* __restrict__ Ap,
                 const unsigned short* __restrict__ Bp,
                 float* __restrict__ C, int Ksteps, int Mtiles) {
  int wave = threadIdx.x >> 5;
  int lane = threadIdx.x & 31;
  int mt = blockIdx.x * 4 + (wave >> 1);
  if (mt >= Mtiles) mt = Mtiles - 1;   // clamp: redundant waves rewrite same tile
  int ntb = (wave & 1) * 4;            // first of 4 n-tiles (64 cols) for this wave

  const unsigned short* a_ptr = Ap + (size_t)mt * Ksteps * 512 + lane * 16;
  const unsigned short* b_ptr = Bp + (size_t)ntb * 512 + lane * 16;

  v8f acc0 = {}, acc1 = {}, acc2 = {}, acc3 = {};
  for (int ks = 0; ks < Ksteps; ++ks) {
    v16bf a  = __builtin_bit_cast(v16bf, *(const v8u*)(a_ptr + (size_t)ks * 512));
    const unsigned short* bk = b_ptr + (size_t)ks * 4096;
    v16bf b0 = __builtin_bit_cast(v16bf, *(const v8u*)(bk));
    v16bf b1 = __builtin_bit_cast(v16bf, *(const v8u*)(bk + 512));
    v16bf b2 = __builtin_bit_cast(v16bf, *(const v8u*)(bk + 1024));
    v16bf b3 = __builtin_bit_cast(v16bf, *(const v8u*)(bk + 1536));
    acc0 = __builtin_amdgcn_wmma_f32_16x16x32_bf16(false, a, false, b0, (short)0, acc0, false, false);
    acc1 = __builtin_amdgcn_wmma_f32_16x16x32_bf16(false, a, false, b1, (short)0, acc1, false, false);
    acc2 = __builtin_amdgcn_wmma_f32_16x16x32_bf16(false, a, false, b2, (short)0, acc2, false, false);
    acc3 = __builtin_amdgcn_wmma_f32_16x16x32_bf16(false, a, false, b3, (short)0, acc3, false, false);
  }

  // C/D layout: elem r -> row (lane>>4)*8 + r, col lane&15
  int n    = lane & 15;
  int mrow = (lane >> 4) * 8;
  float* out = C + (size_t)(mt * 16 + mrow) * 128 + ntb * 16 + n;
#pragma unroll
  for (int r = 0; r < 8; ++r) {
    out[(size_t)r * 128 + 0]  = acc0[r];
    out[(size_t)r * 128 + 16] = acc1[r];
    out[(size_t)r * 128 + 32] = acc2[r];
    out[(size_t)r * 128 + 48] = acc3[r];
  }
}

// el[n,h] = dot(feat[n, h*D : h*D+D], al[h]);  er likewise
__global__ void attn_coeff_kernel(const float* __restrict__ feat,
                                  const float* __restrict__ al,
                                  const float* __restrict__ ar,
                                  float* __restrict__ el, float* __restrict__ er,
                                  int n_nodes, int H, int D) {
  int i = blockIdx.x * blockDim.x + threadIdx.x;
  if (i >= n_nodes * H) return;
  int n = i / H, h = i - n * H;
  const float* f = feat + (size_t)n * H * D + (size_t)h * D;
  float sl = 0.f, sr = 0.f;
  for (int d = 0; d < D; ++d) {
    float v = f[d];
    sl += v * al[h * D + d];
    sr += v * ar[h * D + d];
  }
  el[i] = sl;
  er[i] = sr;
}

__global__ void init_bias_kernel(float* __restrict__ out,
                                 const float* __restrict__ b0,
                                 const float* __restrict__ b1, int total) {
  int i = blockIdx.x * blockDim.x + threadIdx.x;
  if (i >= total) return;
  out[i] = b0[i & 127] + b1[i & 127];
}

__global__ void init_ms_kernel(float* __restrict__ m, float* __restrict__ s, int total) {
  int i = blockIdx.x * blockDim.x + threadIdx.x;
  if (i >= total) return;
  m[i] = __uint_as_float(0xFF800000u);  // -inf
  s[i] = 0.f;
}

// pass1: e = leaky_relu(el[src] + er[dst]); atomic max per dst segment
__global__ void edge_pass1_kernel(const int* __restrict__ src, const int* __restrict__ dst,
                                  const float* __restrict__ el, const float* __restrict__ er,
                                  float* __restrict__ ebuf, float* __restrict__ m,
                                  int E, int H) {
  int e = blockIdx.x * blockDim.x + threadIdx.x;
  if (e >= E) return;
  int u = src[e], v = dst[e];
  for (int h = 0; h < H; ++h) {
    float x = el[u * H + h] + er[v * H + h];
    x = x > 0.f ? x : 0.2f * x;
    ebuf[(size_t)e * H + h] = x;
    atomicMaxF(&m[v * H + h], x);
  }
}

// pass2: ex = exp(e - m[dst]); atomic sum per dst segment
__global__ void edge_pass2_kernel(const int* __restrict__ dst,
                                  float* __restrict__ ebuf,
                                  const float* __restrict__ m, float* __restrict__ s,
                                  int E, int H) {
  int e = blockIdx.x * blockDim.x + threadIdx.x;
  if (e >= E) return;
  int v = dst[e];
  for (int h = 0; h < H; ++h) {
    float ex = __expf(ebuf[(size_t)e * H + h] - m[v * H + h]);
    ebuf[(size_t)e * H + h] = ex;
    atomicAdd(&s[v * H + h], ex);
  }
}

// pass3: out[dst] += feat[src] * (ex / s[dst]); one wave32 per edge, float4/lane
__global__ void edge_pass3_kernel(const int* __restrict__ src, const int* __restrict__ dst,
                                  const float* __restrict__ ebuf, const float* __restrict__ s,
                                  const float* __restrict__ feat, float* __restrict__ out,
                                  int E, int H) {
  int gid = blockIdx.x * blockDim.x + threadIdx.x;
  int e = gid >> 5;
  if (e >= E) return;
  int lane = gid & 31;
  int j0 = lane * 4;                       // 32 lanes x 4 floats = 128 features
  int u = src[e], v = dst[e];
  int h = (H == 1) ? 0 : (j0 >> 5);        // head of this 4-feature slice (D=32)
  float a = ebuf[(size_t)e * H + h] / s[v * H + h];
  float4 f = *(const float4*)(feat + (size_t)u * 128 + j0);
  float* o = out + (size_t)v * 128 + j0;
  atomicAdd(o + 0, f.x * a);
  atomicAdd(o + 1, f.y * a);
  atomicAdd(o + 2, f.z * a);
  atomicAdd(o + 3, f.w * a);
}

__global__ void elu_kernel(float* __restrict__ h, int total) {
  int i = blockIdx.x * blockDim.x + threadIdx.x;
  if (i >= total) return;
  float x = h[i];
  h[i] = x > 0.f ? x : (__expf(x) - 1.f);
}

// ---------------------------------------------------------------------------
extern "C" void kernel_launch(void* const* d_in, const int* in_sizes, int n_in,
                              void* d_out, int out_size, void* d_ws, size_t ws_size,
                              hipStream_t stream) {
  (void)n_in; (void)out_size; (void)ws_size;
  const float* x      = (const float*)d_in[0];
  const float* W1[2]  = {(const float*)d_in[1],  (const float*)d_in[5]};
  const float* al1[2] = {(const float*)d_in[2],  (const float*)d_in[6]};
  const float* ar1[2] = {(const float*)d_in[3],  (const float*)d_in[7]};
  const float* b1[2]  = {(const float*)d_in[4],  (const float*)d_in[8]};
  const float* W2[2]  = {(const float*)d_in[9],  (const float*)d_in[13]};
  const float* al2[2] = {(const float*)d_in[10], (const float*)d_in[14]};
  const float* ar2[2] = {(const float*)d_in[11], (const float*)d_in[15]};
  const float* b2[2]  = {(const float*)d_in[12], (const float*)d_in[16]};
  const int*   src[2] = {(const int*)d_in[17],   (const int*)d_in[19]};
  const int*   dst[2] = {(const int*)d_in[18],   (const int*)d_in[20]};

  const int Nn = in_sizes[0] / 256;   // 50000 (divisible by 16)
  const int E  = in_sizes[17];        // 800000
  const int Mtiles = Nn / 16;         // 3125
  const int Gblocks = (Mtiles + 3) / 4;

  // workspace layout (256B aligned blocks)
  char* ws = (char*)d_ws;
  size_t off = 0;
  auto alloc = [&](size_t bytes) -> void* {
    void* p = ws + off;
    off = (off + bytes + 255) & ~(size_t)255;
    return p;
  };
  unsigned short* xpack = (unsigned short*)alloc((size_t)Nn * 256 * 2); // 25.6 MB
  unsigned short* hpack = (unsigned short*)alloc((size_t)Nn * 128 * 2); // 12.8 MB
  unsigned short* wpack = (unsigned short*)alloc((size_t)65536 * 2);    // packed W
  float* feat = (float*)alloc((size_t)Nn * 128 * 4);                    // 25.6 MB
  float* h1   = (float*)alloc((size_t)Nn * 128 * 4);                    // 25.6 MB
  float* ebuf = (float*)alloc((size_t)E * 4 * 4);                       // 12.8 MB
  float* el   = (float*)alloc((size_t)Nn * 4 * 4);
  float* er   = (float*)alloc((size_t)Nn * 4 * 4);
  float* mbuf = (float*)alloc((size_t)Nn * 4 * 4);
  float* sbuf = (float*)alloc((size_t)Nn * 4 * 4);

  const int B = 256;
  auto grid = [&](long long n) { return dim3((unsigned)((n + B - 1) / B)); };

  // ---- layer 1 ----
  int totA1 = Mtiles * (256 / 32) * 32;
  pack_a_kernel<<<grid(totA1), B, 0, stream>>>(x, xpack, 256, totA1);
  init_bias_kernel<<<grid((long long)Nn * 128), B, 0, stream>>>(h1, b1[0], b1[1], Nn * 128);

  for (int r = 0; r < 2; ++r) {
    int totB = (256 / 32) * 8 * 32;
    pack_b_kernel<<<grid(totB), B, 0, stream>>>(W1[r], wpack, 256, totB);
    gemm_wmma_kernel<<<Gblocks, 256, 0, stream>>>(xpack, wpack, feat, 256 / 32, Mtiles);
    attn_coeff_kernel<<<grid((long long)Nn * 4), B, 0, stream>>>(feat, al1[r], ar1[r],
                                                                 el, er, Nn, 4, 32);
    init_ms_kernel<<<grid((long long)Nn * 4), B, 0, stream>>>(mbuf, sbuf, Nn * 4);
    edge_pass1_kernel<<<grid(E), B, 0, stream>>>(src[r], dst[r], el, er, ebuf, mbuf, E, 4);
    edge_pass2_kernel<<<grid(E), B, 0, stream>>>(dst[r], ebuf, mbuf, sbuf, E, 4);
    edge_pass3_kernel<<<grid((long long)E * 32), B, 0, stream>>>(src[r], dst[r], ebuf, sbuf,
                                                                 feat, h1, E, 4);
  }

  // ---- ELU + repack for layer 2 ----
  elu_kernel<<<grid((long long)Nn * 128), B, 0, stream>>>(h1, Nn * 128);
  int totA2 = Mtiles * (128 / 32) * 32;
  pack_a_kernel<<<grid(totA2), B, 0, stream>>>(h1, hpack, 128, totA2);

  // ---- layer 2 (H=1, D=128; head-mean is identity) ----
  float* outF = (float*)d_out;
  init_bias_kernel<<<grid((long long)Nn * 128), B, 0, stream>>>(outF, b2[0], b2[1], Nn * 128);

  for (int r = 0; r < 2; ++r) {
    int totB = (128 / 32) * 8 * 32;
    pack_b_kernel<<<grid(totB), B, 0, stream>>>(W2[r], wpack, 128, totB);
    gemm_wmma_kernel<<<Gblocks, 256, 0, stream>>>(hpack, wpack, feat, 128 / 32, Mtiles);
    attn_coeff_kernel<<<grid(Nn), B, 0, stream>>>(feat, al2[r], ar2[r], el, er, Nn, 1, 128);
    init_ms_kernel<<<grid(Nn), B, 0, stream>>>(mbuf, sbuf, Nn);
    edge_pass1_kernel<<<grid(E), B, 0, stream>>>(src[r], dst[r], el, er, ebuf, mbuf, E, 1);
    edge_pass2_kernel<<<grid(E), B, 0, stream>>>(dst[r], ebuf, mbuf, sbuf, E, 1);
    edge_pass3_kernel<<<grid((long long)E * 32), B, 0, stream>>>(src[r], dst[r], ebuf, sbuf,
                                                                 feat, outF, E, 1);
  }
}
